// PatchEmbed_59983513256167
// MI455X (gfx1250) — 8-bit matrix operand order repaired and checked by reading, compiled, not run on hardware
//
#include <hip/hip_runtime.h>

typedef __attribute__((ext_vector_type(8))) int v8i;

#define OFF_B 256
#define OFF_A (256 + 147456)

#define M_TILES 3136
#define N_TILES 48
#define K_BLOCKS 3

__global__ void init_scales_kernel(unsigned* s) {
    if (threadIdx.x < 2) s[threadIdx.x] = 0u;
}

__global__ void maxabs4_kernel(const float4* __restrict__ x, long n4, unsigned* __restrict__ dst) {
    __shared__ float red[256];
    float m = 0.f;
    for (long i = (long)blockIdx.x * blockDim.x + threadIdx.x; i < n4;
         i += (long)gridDim.x * blockDim.x) {
        float4 v = x[i];
        m = fmaxf(m, fmaxf(fmaxf(fabsf(v.x), fabsf(v.y)), fmaxf(fabsf(v.z), fabsf(v.w))));
    }
    red[threadIdx.x] = m;
    __syncthreads();
    for (int s2 = 128; s2 > 0; s2 >>= 1) {
        if ((int)threadIdx.x < s2)
            red[threadIdx.x] = fmaxf(red[threadIdx.x], red[threadIdx.x + s2]);
        __syncthreads();
    }
    if (threadIdx.x == 0) atomicMax(dst, __float_as_uint(red[0]));
}

__device__ __forceinline__ unsigned quant4(float4 v, float inv_s) {
    int a = (int)rintf(fminf(fmaxf(v.x * inv_s, -127.f), 127.f));
    int b = (int)rintf(fminf(fmaxf(v.y * inv_s, -127.f), 127.f));
    int c = (int)rintf(fminf(fmaxf(v.z * inv_s, -127.f), 127.f));
    int d = (int)rintf(fminf(fmaxf(v.w * inv_s, -127.f), 127.f));
    return (unsigned)(a & 255) | ((unsigned)(b & 255) << 8) |
           ((unsigned)(c & 255) << 16) | ((unsigned)(d & 255) << 24);
}

__global__ void pack_x_kernel(const float* __restrict__ x, const unsigned* __restrict__ scales,
                              unsigned* __restrict__ dstA) {
    int dw = blockIdx.x * blockDim.x + threadIdx.x;
    float ax = fmaxf(__uint_as_float(scales[0]), 1e-8f);
    float inv_s = 127.f / ax;

    int fragIdx = dw >> 8;
    int rem     = dw & 255;
    int lane    = rem >> 3;
    int v       = rem & 7;
    int half    = lane >> 4;
    int m       = lane & 15;
    int mtile   = fragIdx / 3;
    int c       = fragIdx - mtile * 3;
    int row     = mtile * 16 + m;
    int bb      = row / 784;
    int patch   = row - bb * 784;
    int gx      = patch / 28;
    int gy      = patch - gx * 28;
    int vv      = v & 3;
    int k       = ((v >> 2) << 5) + ((vv >> 1) << 4) + (half << 3) + ((vv & 1) << 2);
    int p       = k >> 3;
    int q       = k & 7;
    long xi = (((long)(bb * 3 + c) * 224) + gx * 8 + p) * 224 + gy * 8 + q;
    float4 xv = *(const float4*)(x + xi);
    dstA[dw] = quant4(xv, inv_s);
}

__global__ void pack_w_kernel(const float* __restrict__ w, const unsigned* __restrict__ scales,
                              unsigned* __restrict__ dstB) {
    int dw = blockIdx.x * blockDim.x + threadIdx.x;
    float aw = fmaxf(__uint_as_float(scales[1]), 1e-8f);
    float inv_s = 127.f / aw;

    int fragIdx = dw >> 8;
    int rem     = dw & 255;
    int lane    = rem >> 3;
    int v       = rem & 7;
    int half    = lane >> 4;
    int nloc    = lane & 15;
    int ntile   = fragIdx / 3;
    int kb      = fragIdx - ntile * 3;
    int n       = ntile * 16 + nloc;
    int k       = ((v >> 2) << 5) + (((v & 3) >> 1) << 4) + (half << 3) + ((v & 1) << 2);
    long wi = (long)n * 192 + kb * 64 + k;
    float4 wv = *(const float4*)(w + wi);
    dstB[dw] = quant4(wv, inv_s);
}

__global__ void gemm_iu8_kernel(const char* __restrict__ wsA, const char* __restrict__ wsB,
                                const unsigned* __restrict__ scales,
                                const float* __restrict__ bias, float* __restrict__ out) {
    int wid  = blockIdx.x * 8 + (threadIdx.x >> 5);
    int lane = threadIdx.x & 31;
    int mtile = wid / 12;
    int ng    = wid - mtile * 12;

    float ax = fmaxf(__uint_as_float(scales[0]), 1e-8f);
    float aw = fmaxf(__uint_as_float(scales[1]), 1e-8f);
    float s  = (ax * aw) * (1.0f / (127.f * 127.f));

    const v8i* A = (const v8i*)wsA;
    const v8i* B = (const v8i*)wsB;

    v8i a0 = A[(mtile * 3 + 0) * 32 + lane];
    v8i a1 = A[(mtile * 3 + 1) * 32 + lane];
    v8i a2 = A[(mtile * 3 + 2) * 32 + lane];

    int half = lane >> 4;
    int nloc = lane & 15;

#pragma unroll
    for (int nt = 0; nt < 4; ++nt) {
        int ntile = ng * 4 + nt;
        v8i b0 = B[(ntile * 3 + 0) * 32 + lane];
        v8i b1 = B[(ntile * 3 + 1) * 32 + lane];
        v8i b2 = B[(ntile * 3 + 2) * 32 + lane];

        v8i acc = {};
        acc = __builtin_amdgcn_wmma_i32_16x16x64_iu8(true, a0, true, b0, acc, false, false);
        acc = __builtin_amdgcn_wmma_i32_16x16x64_iu8(true, a1, true, b1, acc, false, false);
        acc = __builtin_amdgcn_wmma_i32_16x16x64_iu8(true, a2, true, b2, acc, false, false);

        int n = ntile * 16 + nloc;
        float bv = bias[n];
        long base = (long)(mtile * 16 + half * 8) * 768 + n;
#pragma unroll
        for (int v = 0; v < 8; ++v)
            out[base + (long)v * 768] = (float)acc[v] * s + bv;
    }
}

extern "C" void kernel_launch(void* const* d_in, const int* in_sizes, int n_in,
                              void* d_out, int out_size, void* d_ws, size_t ws_size,
                              hipStream_t stream) {
    const float* x    = (const float*)d_in[0];
    const float* w    = (const float*)d_in[1];
    const float* bias = (const float*)d_in[2];
    float* out        = (float*)d_out;

    char* ws = (char*)d_ws;
    unsigned* scales = (unsigned*)ws;
    unsigned* wsB = (unsigned*)(ws + OFF_B);
    unsigned* wsA = (unsigned*)(ws + OFF_A);

    init_scales_kernel<<<1, 32, 0, stream>>>(scales);
    maxabs4_kernel<<<2048, 256, 0, stream>>>((const float4*)x, 9633792 / 4, scales + 0);
    maxabs4_kernel<<<64,   256, 0, stream>>>((const float4*)w, 147456 / 4,  scales + 1);
    pack_x_kernel<<<9408, 256, 0, stream>>>(x, scales, wsA);
    pack_w_kernel<<<144,  256, 0, stream>>>(w, scales, wsB);
    gemm_iu8_kernel<<<4704, 256, 0, stream>>>((const char*)wsA, (const char*)wsB,
                                              scales, bias, out);
}
